// ParallelDroplessMLP_70480413328102
// MI455X (gfx1250) — compile-verified, matrix operand order checked by reading, not run on hardware
//
#include <hip/hip_runtime.h>
#include <hip/hip_bf16.h>

// ---------------- problem constants ----------------
#define SL   2048
#define BSZ  2
#define HS   1024
#define FFN  4096
#define NE   8
#define TOPK 2
#define T    (SL * BSZ)     // 4096 tokens
#define NP   (T * TOPK)     // 8192 (token,expert) pairs

// ---------------- tiling ----------------
#define MTT  128            // rows per block tile
#define NT   128            // cols per block tile
#define KT   32             // K per step (WMMA f16 K)
#define LDSS 40             // LDS row stride in f16 elems (80B: 16B-aligned + pad)

typedef __attribute__((ext_vector_type(16))) _Float16 v16h;
typedef __attribute__((ext_vector_type(8)))  _Float16 v8h;
typedef __attribute__((ext_vector_type(8)))  float    v8f;

__device__ __forceinline__ float gelu_tanh(float v) {
    float u = 0.7978845608028654f * (v + 0.044715f * v * v * v);
    return 0.5f * v * (1.0f + tanhf(u));
}

// ---------------------------------------------------------------------------
// Pass 1: deterministic per-expert token lists (no atomics -> bit-stable).
// ---------------------------------------------------------------------------
__global__ __launch_bounds__(32)
void moe_build(const int* __restrict__ ei,
               int* __restrict__ counts,
               int* __restrict__ offsets,
               int* __restrict__ pair_token,
               int* __restrict__ pair_slot)
{
    __shared__ int shCnt[NE];
    __shared__ int shOff[NE + 1];
    int tid = threadIdx.x;
    if (tid < NE) {
        int c = 0;
        for (int p = 0; p < NP; ++p)
            if (ei[p] == tid) ++c;
        shCnt[tid] = c;
    }
    __syncthreads();
    if (tid == 0) {
        int acc = 0;
        for (int e = 0; e < NE; ++e) { shOff[e] = acc; acc += shCnt[e]; }
        shOff[NE] = acc;
        for (int e = 0; e < NE; ++e) counts[e] = shCnt[e];
        for (int e = 0; e <= NE; ++e) offsets[e] = shOff[e];
    }
    __syncthreads();
    if (tid < NE) {
        int local = 0;
        for (int p = 0; p < NP; ++p) {
            if (ei[p] == tid) {
                int s = shOff[tid] + local;
                pair_token[s] = p >> 1;   // TOPK == 2
                pair_slot[p]  = s;
                ++local;
            }
        }
    }
}

// ---------------------------------------------------------------------------
// Grouped GEMM1 + GELU:  H[slot, :] = gelu( x[token(slot)] @ w1[e] )
// 256 thr / 8 waves; block tile 128x128x32; wave tile 32x64 (8 WMMA/K-step);
// LDS double-buffered, 1 barrier per K-step, f16 WMMA + f32 accumulate.
// ---------------------------------------------------------------------------
__global__ __launch_bounds__(256)
void moe_gemm1_gelu(const float* __restrict__ x,          // [T, HS]
                    const float* __restrict__ w1,         // [NE, HS, FFN]
                    const int*   __restrict__ pair_token, // [NP]
                    const int*   __restrict__ offsets,    // [NE+1]
                    _Float16*    __restrict__ H)          // [NP, FFN]
{
    const int e    = blockIdx.z;
    const int rbeg = offsets[e];
    const int rows = offsets[e + 1] - rbeg;
    const int mt   = blockIdx.y;
    if (mt * MTT >= rows) return;             // uniform exit: EXEC stays full
    const int nt   = blockIdx.x;

    __shared__ _Float16 Atile[2][MTT * LDSS];
    __shared__ _Float16 Btile[2][NT * LDSS];
    __shared__ int      tokLds[MTT];

    const int tid = threadIdx.x;
    if (tid < MTT) {
        int r = mt * MTT + tid;
        tokLds[tid] = (r < rows) ? pair_token[rbeg + r] : -1;
    }
    __syncthreads();

    const float* w1e = w1 + (size_t)e * HS * FFN;

    // A loader: 2 threads per row, 16 consecutive k each
    const int arow = tid >> 1;
    const int ak0  = (tid & 1) * 16;
    const int atok = tokLds[arow];
    // B loader: one column each, 16 consecutive k each
    const int bn   = tid & (NT - 1);
    const int bk0  = (tid >> 7) * 16;
    const int ncol = nt * NT + bn;

    const int wave  = tid >> 5;
    const int lane  = tid & 31;
    const int wm    = wave & 3;               // 4 wave-rows of 32
    const int wn    = wave >> 2;              // 2 wave-cols of 64
    const int mlane = lane & 15;
    const int hi    = lane >> 4;

    float4 ar[4];
    float  br[16];

    auto loadG = [&](int kb) {
        if (atok >= 0) {
            const float4* src = (const float4*)(x + (size_t)atok * HS + kb + ak0);
            #pragma unroll
            for (int q = 0; q < 4; ++q) ar[q] = src[q];
        } else {
            #pragma unroll
            for (int q = 0; q < 4; ++q) ar[q] = make_float4(0.f, 0.f, 0.f, 0.f);
        }
        const float* bsrc = w1e + (size_t)(kb + bk0) * FFN + ncol;
        #pragma unroll
        for (int j = 0; j < 16; ++j) br[j] = bsrc[(size_t)j * FFN];
    };
    auto storeL = [&](int b) {
        v8h a0, a1;
        #pragma unroll
        for (int q = 0; q < 2; ++q) {
            a0[4*q+0] = (_Float16)ar[q].x;   a0[4*q+1] = (_Float16)ar[q].y;
            a0[4*q+2] = (_Float16)ar[q].z;   a0[4*q+3] = (_Float16)ar[q].w;
            a1[4*q+0] = (_Float16)ar[q+2].x; a1[4*q+1] = (_Float16)ar[q+2].y;
            a1[4*q+2] = (_Float16)ar[q+2].z; a1[4*q+3] = (_Float16)ar[q+2].w;
        }
        *(v8h*)&Atile[b][arow * LDSS + ak0]     = a0;
        *(v8h*)&Atile[b][arow * LDSS + ak0 + 8] = a1;
        v8h b0, b1;
        #pragma unroll
        for (int j = 0; j < 8; ++j) { b0[j] = (_Float16)br[j]; b1[j] = (_Float16)br[j + 8]; }
        *(v8h*)&Btile[b][bn * LDSS + bk0]     = b0;
        *(v8h*)&Btile[b][bn * LDSS + bk0 + 8] = b1;
    };

    v8f acc[2][4] = {};

    loadG(0);
    storeL(0);
    __syncthreads();

    int buf = 0;
    for (int kb = 0;;) {
        int kn = kb + KT;
        if (kn < HS) loadG(kn);               // prefetch next slice into regs

        // ---- load ALL fragments, then issue the WMMA block ----
        v16h afrag[2];
        #pragma unroll
        for (int i = 0; i < 2; ++i) {
            int row = wm * 32 + i * 16 + mlane;
            v8h lo = *(const v8h*)&Atile[buf][row * LDSS + 8 * hi];
            v8h hh = *(const v8h*)&Atile[buf][row * LDSS + 16 + 8 * hi];
            #pragma unroll
            for (int q = 0; q < 8; ++q) { afrag[i][q] = lo[q]; afrag[i][q + 8] = hh[q]; }
        }
        v16h bfrag[4];
        #pragma unroll
        for (int j = 0; j < 4; ++j) {
            int n = wn * 64 + j * 16 + mlane;
            v8h lo = *(const v8h*)&Btile[buf][n * LDSS + 8 * hi];
            v8h hh = *(const v8h*)&Btile[buf][n * LDSS + 16 + 8 * hi];
            #pragma unroll
            for (int q = 0; q < 8; ++q) { bfrag[j][q] = lo[q]; bfrag[j][q + 8] = hh[q]; }
        }
        #pragma unroll
        for (int i = 0; i < 2; ++i)
            #pragma unroll
            for (int j = 0; j < 4; ++j)
                acc[i][j] = __builtin_amdgcn_wmma_f32_16x16x32_f16(
                    false, afrag[i], false, bfrag[j], (short)0, acc[i][j], false, false);

        if (kn >= HS) break;
        storeL(buf ^ 1);                      // other buffer: no conflict w/ readers
        __syncthreads();
        buf ^= 1;
        kb = kn;
    }

    // ---- epilogue: GELU, f16 store. C/D layout: VGPR v -> M = v + 8*hi ----
    #pragma unroll
    for (int i = 0; i < 2; ++i) {
        #pragma unroll
        for (int j = 0; j < 4; ++j) {
            int n_g = nt * NT + wn * 64 + j * 16 + mlane;
            #pragma unroll
            for (int v = 0; v < 8; ++v) {
                int mrow = wm * 32 + i * 16 + v + 8 * hi;
                if (mt * MTT + mrow < rows) {
                    size_t slot = (size_t)(rbeg + mt * MTT + mrow);
                    H[slot * FFN + n_g] = (_Float16)gelu_tanh(acc[i][j][v]);
                }
            }
        }
    }
}

// ---------------------------------------------------------------------------
// Grouped GEMM2:  Y[slot, :] = H[slot] @ w2[e]   (f32 out)
// ---------------------------------------------------------------------------
__global__ __launch_bounds__(256)
void moe_gemm2(const _Float16* __restrict__ H,        // [NP, FFN] f16
               const float*    __restrict__ w2,       // [NE, FFN, HS]
               const int*      __restrict__ offsets,  // [NE+1]
               float*          __restrict__ Y)        // [NP, HS]
{
    const int e    = blockIdx.z;
    const int rbeg = offsets[e];
    const int rows = offsets[e + 1] - rbeg;
    const int mt   = blockIdx.y;
    if (mt * MTT >= rows) return;
    const int nt   = blockIdx.x;

    __shared__ _Float16 Atile[2][MTT * LDSS];
    __shared__ _Float16 Btile[2][NT * LDSS];

    const int tid = threadIdx.x;
    const float* w2e = w2 + (size_t)e * FFN * HS;

    const int arow = tid >> 1;
    const int ak0  = (tid & 1) * 16;
    const int bn   = tid & (NT - 1);
    const int bk0  = (tid >> 7) * 16;
    const int ncol = nt * NT + bn;

    const int wave  = tid >> 5;
    const int lane  = tid & 31;
    const int wm    = wave & 3;
    const int wn    = wave >> 2;
    const int mlane = lane & 15;
    const int hi    = lane >> 4;

    const bool arow_ok = (mt * MTT + arow) < rows;
    const size_t aslot = (size_t)(rbeg + mt * MTT + arow);

    v8h   ar2[2];
    float br[16];

    auto loadG = [&](int kb) {
        if (arow_ok) {
            const v8h* src = (const v8h*)(H + aslot * FFN + kb + ak0);
            ar2[0] = src[0];
            ar2[1] = src[1];
        } else {
            #pragma unroll
            for (int q = 0; q < 8; ++q) { ar2[0][q] = (_Float16)0.f; ar2[1][q] = (_Float16)0.f; }
        }
        const float* bsrc = w2e + (size_t)(kb + bk0) * HS + ncol;
        #pragma unroll
        for (int j = 0; j < 16; ++j) br[j] = bsrc[(size_t)j * HS];
    };
    auto storeL = [&](int b) {
        *(v8h*)&Atile[b][arow * LDSS + ak0]     = ar2[0];
        *(v8h*)&Atile[b][arow * LDSS + ak0 + 8] = ar2[1];
        v8h b0, b1;
        #pragma unroll
        for (int j = 0; j < 8; ++j) { b0[j] = (_Float16)br[j]; b1[j] = (_Float16)br[j + 8]; }
        *(v8h*)&Btile[b][bn * LDSS + bk0]     = b0;
        *(v8h*)&Btile[b][bn * LDSS + bk0 + 8] = b1;
    };

    v8f acc[2][4] = {};

    loadG(0);
    storeL(0);
    __syncthreads();

    int buf = 0;
    for (int kb = 0;;) {
        int kn = kb + KT;
        if (kn < FFN) loadG(kn);

        v16h afrag[2];
        #pragma unroll
        for (int i = 0; i < 2; ++i) {
            int row = wm * 32 + i * 16 + mlane;
            v8h lo = *(const v8h*)&Atile[buf][row * LDSS + 8 * hi];
            v8h hh = *(const v8h*)&Atile[buf][row * LDSS + 16 + 8 * hi];
            #pragma unroll
            for (int q = 0; q < 8; ++q) { afrag[i][q] = lo[q]; afrag[i][q + 8] = hh[q]; }
        }
        v16h bfrag[4];
        #pragma unroll
        for (int j = 0; j < 4; ++j) {
            int n = wn * 64 + j * 16 + mlane;
            v8h lo = *(const v8h*)&Btile[buf][n * LDSS + 8 * hi];
            v8h hh = *(const v8h*)&Btile[buf][n * LDSS + 16 + 8 * hi];
            #pragma unroll
            for (int q = 0; q < 8; ++q) { bfrag[j][q] = lo[q]; bfrag[j][q + 8] = hh[q]; }
        }
        #pragma unroll
        for (int i = 0; i < 2; ++i)
            #pragma unroll
            for (int j = 0; j < 4; ++j)
                acc[i][j] = __builtin_amdgcn_wmma_f32_16x16x32_f16(
                    false, afrag[i], false, bfrag[j], (short)0, acc[i][j], false, false);

        if (kn >= FFN) break;
        storeL(buf ^ 1);
        __syncthreads();
        buf ^= 1;
        kb = kn;
    }

    #pragma unroll
    for (int i = 0; i < 2; ++i) {
        #pragma unroll
        for (int j = 0; j < 4; ++j) {
            int n_g = nt * NT + wn * 64 + j * 16 + mlane;
            #pragma unroll
            for (int v = 0; v < 8; ++v) {
                int mrow = wm * 32 + i * 16 + v + 8 * hi;
                if (mt * MTT + mrow < rows) {
                    size_t slot = (size_t)(rbeg + mt * MTT + mrow);
                    Y[slot * HS + n_g] = acc[i][j][v];
                }
            }
        }
    }
}

// ---------------------------------------------------------------------------
// Combine: out[t] = ew[t,0]*Y[slot(t,0)] + ew[t,1]*Y[slot(t,1)]
// ---------------------------------------------------------------------------
__global__ __launch_bounds__(256)
void moe_combine(const float* __restrict__ ew,
                 const int*   __restrict__ pair_slot,
                 const float* __restrict__ Y,
                 float*       __restrict__ out)
{
    int idx = blockIdx.x * blockDim.x + threadIdx.x;
    if (idx >= T * HS) return;
    int t = idx >> 10;          // HS == 1024
    int n = idx & (HS - 1);
    int s0 = pair_slot[2 * t];
    int s1 = pair_slot[2 * t + 1];
    out[idx] = ew[2 * t]     * Y[(size_t)s0 * HS + n]
             + ew[2 * t + 1] * Y[(size_t)s1 * HS + n];
}

// ---------------------------------------------------------------------------
extern "C" void kernel_launch(void* const* d_in, const int* in_sizes, int n_in,
                              void* d_out, int out_size, void* d_ws, size_t ws_size,
                              hipStream_t stream) {
    const float* x  = (const float*)d_in[0];
    const float* ew = (const float*)d_in[1];
    const int*   ei = (const int*)  d_in[2];
    const float* w1 = (const float*)d_in[3];
    const float* w2 = (const float*)d_in[4];
    float* out = (float*)d_out;

    char* ws = (char*)d_ws;
    int* counts     = (int*)(ws);
    int* offsets    = (int*)(ws + 64);
    int* pair_token = (int*)(ws + 128);
    int* pair_slot  = (int*)(ws + 128 + (size_t)NP * 4);
    _Float16* H     = (_Float16*)(ws + (1u << 20));                          // 64 MB
    float*    Y     = (float*)(ws + (1u << 20) + (size_t)NP * FFN * 2);      // 32 MB

    moe_build<<<1, 32, 0, stream>>>(ei, counts, offsets, pair_token, pair_slot);

    dim3 g1(FFN / NT, NP / MTT, NE);   // 32 x 64 x 8
    moe_gemm1_gelu<<<g1, 256, 0, stream>>>(x, w1, pair_token, offsets, H);

    dim3 g2(HS / NT, NP / MTT, NE);    // 8 x 64 x 8
    moe_gemm2<<<g2, 256, 0, stream>>>(H, w2, offsets, Y);

    moe_combine<<<(T * HS + 255) / 256, 256, 0, stream>>>(ew, pair_slot, Y, out);
}
